// AttentionBlock_16587163697991
// MI455X (gfx1250) — matrix operand order repaired and checked by reading, not compiled, not run on hardware
//
#include <hip/hip_runtime.h>
#include <hip/hip_bf16.h>
#include <cstdint>

typedef __attribute__((ext_vector_type(16))) _Float16 v16h;
typedef __attribute__((ext_vector_type(8)))  _Float16 v8h;
typedef __attribute__((ext_vector_type(8)))  float    v8f;

#define CH 512
#define HEADS 4
#define HD 128
#define NSEQ 1024
#define NB 16
#define GROUPS 32
#define QSCALE 0.08838834764831845f   // 128^-0.5

// ---------------- WMMA + wide fragment loaders (wave32, 16x16x32 f16) --------

__device__ inline v8f wmma_f16(v16h a, v16h b, v8f c) {
    return __builtin_amdgcn_wmma_f32_16x16x32_f16(false, a, false, b, (short)0, c, false, false);
}

__device__ inline v16h cat8(v8h lo, v8h hi) {
    return __builtin_shufflevector(lo, hi, 0,1,2,3,4,5,6,7,8,9,10,11,12,13,14,15);
}

// A fragment 16(M) x 32(K): source row-major [M][K], ld halfs (ld*2 % 16 == 0).
// lane<16: K = 0..15; lane>=16: K = 16..31  -> two 16B loads.
__device__ inline v16h fragA_ld(const _Float16* base, int ld) {
    int lane = threadIdx.x & 31;
    const _Float16* p = base + (size_t)(lane & 15) * ld + ((lane < 16) ? 0 : 16);
    return cat8(*(const v8h*)p, *(const v8h*)(p + 8));
}

// B fragment 32(K) x 16(N): source stored [N][K], ld halfs.
// lane<16: K = 0..15; lane>=16: K = 16..31  -> two 16B loads.
__device__ inline v16h fragB_ld(const _Float16* base, int ld) {
    int lane = threadIdx.x & 31;
    const _Float16* p = base + (size_t)(lane & 15) * ld + ((lane < 16) ? 0 : 16);
    return cat8(*(const v8h*)p, *(const v8h*)(p + 8));
}

// ---------------- GroupNorm: f32 [b][c][n] -> f16 xn_t [b][n][c] -------------

__global__ void __launch_bounds__(256) gn_kernel(const float* __restrict__ x,
                                                 const float* __restrict__ gamma,
                                                 const float* __restrict__ beta,
                                                 _Float16* __restrict__ xnt) {
    const int tid = threadIdx.x;
    const int bb = blockIdx.x >> 5;
    const int g  = blockIdx.x & 31;
    const float* xb = x + ((size_t)bb * CH + g * 16) * NSEQ;

    float s = 0.f, s2 = 0.f;
    for (int i = tid; i < 16 * NSEQ; i += 256) {
        float v = xb[i];
        s += v; s2 += v * v;
    }
    __shared__ float rs[256], rs2[256];
    rs[tid] = s; rs2[tid] = s2;
    __syncthreads();
    for (int st = 128; st > 0; st >>= 1) {
        if (tid < st) { rs[tid] += rs[tid + st]; rs2[tid] += rs2[tid + st]; }
        __syncthreads();
    }
    const float inv = 1.f / (16.f * NSEQ);
    float mu   = rs[0] * inv;
    float var  = rs2[0] * inv - mu * mu;
    float rstd = rsqrtf(var + 1e-5f);

    __shared__ alignas(16) _Float16 T[16 * NSEQ];   // [n][cl], 32 KB
    for (int i = tid; i < 16 * NSEQ; i += 256) {
        int cl = i >> 10, n = i & (NSEQ - 1);
        int c = g * 16 + cl;
        T[n * 16 + cl] = (_Float16)((xb[i] - mu) * rstd * gamma[c] + beta[c]);
    }
    __syncthreads();
    for (int n = tid; n < NSEQ; n += 256) {
        _Float16* dst = xnt + ((size_t)bb * NSEQ + n) * CH + g * 16;
        *(v8h*)dst       = *(const v8h*)(T + n * 16);
        *(v8h*)(dst + 8) = *(const v8h*)(T + n * 16 + 8);
    }
}

// ---------------- weight conversion f32 -> f16 -------------------------------

__global__ void __launch_bounds__(256) cvt_kernel(const float* __restrict__ wq,
                                                  const float* __restrict__ wp,
                                                  _Float16* __restrict__ wqh,
                                                  _Float16* __restrict__ wph) {
    int i = blockIdx.x * 256 + threadIdx.x;
    if (i < 3 * CH * CH) wqh[i] = (_Float16)wq[i];
    if (i < CH * CH)     wph[i] = (_Float16)wp[i];
}

// ---------------- tiled WMMA GEMM: out[m][n] = W[m][:] . Xt[n][:] ------------
// X is stored K-major: Xt[b][n][K]. MODE 0: QKV (half out, q/k transposed to
// qkt[b][n][m], v row-major vh[b][c][n], +bias, q scaled). MODE 1: proj
// (f32 out = acc + bias + residual).

template<int MODE>
__global__ void __launch_bounds__(256) gemm_wmma(const _Float16* __restrict__ W,    // [M][K]
                                                 const _Float16* __restrict__ Xt,   // [B][NSEQ][K]
                                                 const float* __restrict__ bias,    // [M]
                                                 const float* __restrict__ resid,   // [B][M][NSEQ]
                                                 _Float16* __restrict__ out_qk,     // [B][NSEQ][1024]
                                                 _Float16* __restrict__ out_v,      // [B][CH][NSEQ]
                                                 float* __restrict__ out_f,         // [B][M][NSEQ]
                                                 int M, int K) {
    const int tid  = threadIdx.x;
    const int wv   = tid >> 5;
    const int lane = tid & 31;
    const int m0 = blockIdx.x * 128;
    const int n0 = blockIdx.y * 128;
    const int b  = blockIdx.z;
    const _Float16* Xb = Xt + (size_t)b * NSEQ * K;

    const int mbase = (wv & 3) * 32;
    const int nbase = (wv >> 2) * 64;

    __shared__ alignas(16) _Float16 As[128 * 32];   // [m][k]
    __shared__ alignas(16) _Float16 Bs[128 * 32];   // [n][k]  (K-major!)

    v8f acc[2][4];
#pragma unroll
    for (int i = 0; i < 2; ++i)
#pragma unroll
        for (int j = 0; j < 4; ++j) acc[i][j] = {};

    for (int kk = 0; kk < K; kk += 32) {
        for (int i = tid; i < 512; i += 256) {
            int r = i >> 2, c = i & 3;
            ((uint4*)As)[i] = ((const uint4*)(W  + (size_t)(m0 + r) * K + kk))[c];
        }
        for (int i = tid; i < 512; i += 256) {
            int r = i >> 2, c = i & 3;
            ((uint4*)Bs)[i] = ((const uint4*)(Xb + (size_t)(n0 + r) * K + kk))[c];
        }
        __syncthreads();

        v16h af0 = fragA_ld(As + (size_t)mbase * 32, 32);
        v16h af1 = fragA_ld(As + (size_t)(mbase + 16) * 32, 32);
#pragma unroll
        for (int j = 0; j < 4; ++j) {
            v16h bf = fragB_ld(Bs + (size_t)(nbase + 16 * j) * 32, 32);
            acc[0][j] = wmma_f16(af0, bf, acc[0][j]);
            acc[1][j] = wmma_f16(af1, bf, acc[1][j]);
        }
        __syncthreads();
    }

    // C layout: lane holds column n = lane&15; vgpr jj holds row mb+jj where
    // mb = tile_m + 8*(lane>=16). The 8 rows are contiguous -> packed stores.
    const int nl = lane & 15;
    const int mh = (lane < 16) ? 0 : 8;
#pragma unroll
    for (int i = 0; i < 2; ++i) {
        int mb = m0 + mbase + 16 * i + mh;
#pragma unroll
        for (int j = 0; j < 4; ++j) {
            int n = n0 + nbase + 16 * j + nl;
            if (MODE == 0) {
                if (mb < 2 * CH) {          // q or k -> transposed, packed
                    v8h pk;
#pragma unroll
                    for (int jj = 0; jj < 8; ++jj) {
                        float v = acc[i][j][jj] + bias[mb + jj];
                        if (mb < CH) v *= QSCALE;
                        pk[jj] = (_Float16)v;
                    }
                    *(v8h*)(out_qk + ((size_t)b * NSEQ + n) * 1024 + mb) = pk;
                } else {                    // v -> row-major [c][n]
#pragma unroll
                    for (int jj = 0; jj < 8; ++jj) {
                        float v = acc[i][j][jj] + bias[mb + jj];
                        out_v[((size_t)b * CH + (mb + jj - 2 * CH)) * NSEQ + n] = (_Float16)v;
                    }
                }
            } else {
#pragma unroll
                for (int jj = 0; jj < 8; ++jj) {
                    size_t idx = (size_t)b * M * NSEQ + (size_t)(mb + jj) * NSEQ + n;
                    out_f[idx] = acc[i][j][jj] + bias[mb + jj] + resid[idx];
                }
            }
        }
    }
}

// ---------------- flash attention, S^T / O^T orientation, no LDS -------------
// qkt[b][n][0:512]=q (pre-scaled), [512:1024]=k ; vh[b][c][n].
// S^T tile: M=keys, N=queries -> softmax stats are per-lane scalars.
// O^T = V * P^T : A=V rows (wide global), B=P^T built with shfl_xor(16).

__global__ void __launch_bounds__(256) attn_wmma(const _Float16* __restrict__ qkt,
                                                 const _Float16* __restrict__ vh,
                                                 _Float16* __restrict__ att_t) {
    const int tid  = threadIdx.x;
    const int wv   = tid >> 5;
    const int lane = tid & 31;
    const int b = blockIdx.z;
    const int h = blockIdx.y;
    const int q0 = blockIdx.x * 128 + wv * 16;
    const bool hi = lane >= 16;

    const _Float16* Qb = qkt + (size_t)b * NSEQ * 1024 + h * HD;        // + q*1024
    const _Float16* Kb = Qb + CH;
    const _Float16* Vb = vh  + ((size_t)b * CH + h * HD) * NSEQ;        // [c][n]

    // Q as B-fragments: B(k=c, n=query), source rows [q][c]
    v16h bq[4];
#pragma unroll
    for (int cc = 0; cc < 4; ++cc)
        bq[cc] = fragB_ld(Qb + (size_t)q0 * 1024 + cc * 32, 1024);

    v8f o[8];
#pragma unroll
    for (int t = 0; t < 8; ++t) o[t] = {};
    float mrun = -1e30f, lrun = 0.f;

    for (int m0 = 0; m0 < NSEQ; m0 += 32) {
        // S^T: two 16x16 tiles (keys m0..15, m0+16..31) x (16 queries)
        v8f st0 = {}, st1 = {};
#pragma unroll
        for (int cc = 0; cc < 4; ++cc) {
            v16h ak0 = fragA_ld(Kb + (size_t)m0 * 1024 + cc * 32, 1024);
            v16h ak1 = fragA_ld(Kb + (size_t)(m0 + 16) * 1024 + cc * 32, 1024);
            st0 = wmma_f16(ak0, bq[cc], st0);
            st1 = wmma_f16(ak1, bq[cc], st1);
        }

        // online softmax: lane = query; its 16 key-values live in vgprs,
        // partner lane (^16) holds the other 16 keys of the chunk.
        float cmax = fmaxf(st0[0], st1[0]);
#pragma unroll
        for (int j = 1; j < 8; ++j) cmax = fmaxf(cmax, fmaxf(st0[j], st1[j]));
        cmax = fmaxf(cmax, __shfl_xor(cmax, 16, 32));
        float nm = fmaxf(mrun, cmax);
        float r  = __expf(mrun - nm);
        float e0[8], e1[8], rsum = 0.f;
#pragma unroll
        for (int j = 0; j < 8; ++j) {
            e0[j] = __expf(st0[j] - nm);
            e1[j] = __expf(st1[j] - nm);
            rsum += e0[j] + e1[j];
        }
        rsum += __shfl_xor(rsum, 16, 32);
        lrun = lrun * r + rsum;
        mrun = nm;
#pragma unroll
        for (int t = 0; t < 8; ++t)
#pragma unroll
            for (int j = 0; j < 8; ++j) o[t][j] *= r;

        // P^T B-fragment: lane=query, K=key. lane<16 needs keys {0..15},
        // lane>=16 needs keys {16..31}; swap halves with partner lane.
        v16h pb;
#pragma unroll
        for (int j = 0; j < 8; ++j) {
            float s0x = __shfl_xor(e0[j], 16, 32);
            float s1x = __shfl_xor(e1[j], 16, 32);
            pb[j]     = (_Float16)(hi ? s1x : e0[j]);
            pb[j + 8] = (_Float16)(hi ? e1[j] : s0x);
        }

        // O^T += V(16c x 32m) * P^T(32m x 16q), 8 channel tiles
#pragma unroll
        for (int ct = 0; ct < 8; ++ct) {
            v16h av = fragA_ld(Vb + (size_t)(ct * 16) * NSEQ + m0, NSEQ);
            o[ct] = wmma_f16(av, pb, o[ct]);
        }
    }

    // epilogue: lane = query col, vgpr j = channel row (contiguous) -> packed
    float linv = 1.f / lrun;
    _Float16* Ob = att_t + ((size_t)b * NSEQ + q0 + (lane & 15)) * CH
                 + h * HD + (hi ? 8 : 0);
#pragma unroll
    for (int ct = 0; ct < 8; ++ct) {
        v8h pk;
#pragma unroll
        for (int j = 0; j < 8; ++j) pk[j] = (_Float16)(o[ct][j] * linv);
        *(v8h*)(Ob + ct * 16) = pk;
    }
}

// ---------------- host launcher ----------------------------------------------

extern "C" void kernel_launch(void* const* d_in, const int* in_sizes, int n_in,
                              void* d_out, int out_size, void* d_ws, size_t ws_size,
                              hipStream_t stream) {
    (void)in_sizes; (void)n_in; (void)out_size; (void)ws_size;
    const float* x      = (const float*)d_in[0];
    const float* gamma  = (const float*)d_in[1];
    const float* beta   = (const float*)d_in[2];
    const float* w_qkv  = (const float*)d_in[3];
    const float* b_qkv  = (const float*)d_in[4];
    const float* w_proj = (const float*)d_in[5];
    const float* b_proj = (const float*)d_in[6];

    char* ws = (char*)d_ws;
    _Float16* xnt  = (_Float16*)(ws + 0);          // 16 MB : [B][1024][512]
    _Float16* qkt  = (_Float16*)(ws + 16777216);   // 32 MB : [B][1024][1024] (q|k, n-major)
    _Float16* vh   = (_Float16*)(ws + 50331648);   // 16 MB : [B][512][1024]
    _Float16* attt = (_Float16*)(ws + 67108864);   // 16 MB : [B][1024][512]
    _Float16* wqh  = (_Float16*)(ws + 83886080);   // 1.5 MB
    _Float16* wph  = (_Float16*)(ws + 85458944);   // 0.5 MB

    cvt_kernel<<<3072, 256, 0, stream>>>(w_qkv, w_proj, wqh, wph);
    gn_kernel<<<NB * GROUPS, 256, 0, stream>>>(x, gamma, beta, xnt);
    gemm_wmma<0><<<dim3(12, 8, NB), 256, 0, stream>>>(wqh, xnt, b_qkv, nullptr,
                                                      qkt, vh, nullptr, 3 * CH, CH);
    attn_wmma<<<dim3(8, HEADS, NB), 256, 0, stream>>>(qkt, vh, attt);
    gemm_wmma<1><<<dim3(4, 8, NB), 256, 0, stream>>>(wph, attt, b_proj, x,
                                                     nullptr, nullptr, (float*)d_out,
                                                     CH, CH);
}
